// MultiHeadAttention_45372034515280
// MI455X (gfx1250) — compile-verified, matrix operand order checked by reading, not run on hardware
//
#include <hip/hip_runtime.h>

// ---------------- Types for CDNA5 WMMA ----------------
typedef __attribute__((ext_vector_type(16))) __bf16 v16bf;  // 8 VGPRs: A/B operand of wmma_f32_16x16x32_bf16
typedef __attribute__((ext_vector_type(8)))  float  v8f;    // 8 VGPRs: C/D 16x16 f32 accumulator

union V16U {        // reinterpret two B128 loads as one WMMA operand
    v16bf v;
    uint4 u[2];
};

#define BH    32      // B*H
#define SEQN  2048    // N == M
#define DH    64      // head dim
#define SCALE 8.0f

// ---------------- Kernel 1: zero global colsum accumulator ----------------
__global__ void zero_colsum_kernel(float* __restrict__ col) {
    int i = blockIdx.x * blockDim.x + threadIdx.x;
    col[i] = 0.0f;
}

// ---------------- Kernel 2: fp32 -> bf16 conversion of Q and K ----------------
__device__ __forceinline__ uint2 pack4_bf16(float4 a) {
    union { __bf16 h[4]; uint2 u; } p;
    p.h[0] = (__bf16)a.x; p.h[1] = (__bf16)a.y;
    p.h[2] = (__bf16)a.z; p.h[3] = (__bf16)a.w;
    return p.u;
}

__global__ void cvt_bf16_kernel(const float* __restrict__ q, const float* __restrict__ k,
                                unsigned short* __restrict__ qb, unsigned short* __restrict__ kb) {
    int i = blockIdx.x * blockDim.x + threadIdx.x;  // float4 index
    float4 a = ((const float4*)q)[i];
    float4 b = ((const float4*)k)[i];
    ((uint2*)qb)[i] = pack4_bf16(a);
    ((uint2*)kb)[i] = pack4_bf16(b);
}

// ---------------- Score-tile helper: one 16-col key tile vs 32 query rows ----------------
// B = K^T : B[k][n] = K[t*16+n][k] -> per-lane gather from row-major K is identical to the
// A-operand gather (ISA 16-bit 16x32 layout): half 0 holds K 0..7 / 16..23, half 1 K 8..15 / 24..31.
__device__ __forceinline__ void score_tiles(const __bf16* __restrict__ kbase, int t, int lr, int half,
                                            const v16bf& a0L, const v16bf& a1L,
                                            const v16bf& a0H, const v16bf& a1H,
                                            v8f& acc0, v8f& acc1) {
    const uint4* krow = (const uint4*)(kbase + (size_t)(t * 16 + lr) * DH);
    V16U b0, b1;
    b0.u[0] = krow[0 + half];  b0.u[1] = krow[2 + half];   // d 0..31 chunk
    b1.u[0] = krow[4 + half];  b1.u[1] = krow[6 + half];   // d 32..63 chunk
    const v8f z = {0.f, 0.f, 0.f, 0.f, 0.f, 0.f, 0.f, 0.f};
    acc0 = __builtin_amdgcn_wmma_f32_16x16x32_bf16(false, a0L, false, b0.v, (short)0, z,    false, false);
    acc0 = __builtin_amdgcn_wmma_f32_16x16x32_bf16(false, a1L, false, b1.v, (short)0, acc0, false, false);
    acc1 = __builtin_amdgcn_wmma_f32_16x16x32_bf16(false, a0H, false, b0.v, (short)0, z,    false, false);
    acc1 = __builtin_amdgcn_wmma_f32_16x16x32_bf16(false, a1H, false, b1.v, (short)0, acc1, false, false);
}

// ---------------- Kernel 3: WMMA scores + 3-pass softmax + column-sum ----------------
// grid.x = BH * 16 ; block = 128 threads = 4 wave32; each wave owns 32 query rows (two A tiles
// sharing every B tile -> 4 WMMAs per B gather).
__launch_bounds__(128)
__global__ void attn_colsum_kernel(const unsigned short* __restrict__ qb_u,
                                   const unsigned short* __restrict__ kb_u,
                                   float* __restrict__ gcol) {
    __shared__ float scol[SEQN];   // 8 KB of the 320 KB/WGP LDS

    const __bf16* qb = (const __bf16*)qb_u;
    const __bf16* kb = (const __bf16*)kb_u;

    const int bh   = blockIdx.x >> 4;   // head index 0..31
    const int rg   = blockIdx.x & 15;   // 128-row group within head
    const int wave = threadIdx.x >> 5;  // 0..3
    const int lane = threadIdx.x & 31;
    const int half = lane >> 4;         // C layout: vgpr j <-> row j + 8*half, lane%16 <-> column
    const int lr   = lane & 15;

    for (int i = threadIdx.x; i < SEQN; i += 128) scol[i] = 0.0f;
    __syncthreads();

    const int    n0   = rg * 128 + wave * 32;       // first query row of this wave
    const size_t base = (size_t)bh * SEQN * DH;

    // ---- A operands: Q rows n0..n0+31, d=0..63 (4x B128 per 16-row tile) ----
    const uint4* qrowL = (const uint4*)(qb + base + (size_t)(n0 + lr) * DH);
    const uint4* qrowH = (const uint4*)(qb + base + (size_t)(n0 + 16 + lr) * DH);
    V16U a0L, a1L, a0H, a1H;
    a0L.u[0] = qrowL[0 + half];  a0L.u[1] = qrowL[2 + half];
    a1L.u[0] = qrowL[4 + half];  a1L.u[1] = qrowL[6 + half];
    a0H.u[0] = qrowH[0 + half];  a0H.u[1] = qrowH[2 + half];
    a1H.u[0] = qrowH[4 + half];  a1H.u[1] = qrowH[6 + half];

    const __bf16* kbase = kb + base;
    v8f acc0, acc1;

    // ---- Pass 1: raw per-lane max (no exp; SCALE>0 is monotone, fold it later) ----
    float mx[16];
    #pragma unroll
    for (int j = 0; j < 16; ++j) mx[j] = -1e30f;

    for (int t = 0; t < SEQN / 16; ++t) {
        score_tiles(kbase, t, lr, half, a0L.v, a1L.v, a0H.v, a1H.v, acc0, acc1);
        #pragma unroll
        for (int j = 0; j < 8; ++j) {
            mx[j]     = fmaxf(mx[j],     acc0[j]);
            mx[8 + j] = fmaxf(mx[8 + j], acc1[j]);
        }
    }
    // exact row max: reduce across the 16 column-lanes of each half
    #pragma unroll
    for (int off = 1; off <= 8; off <<= 1) {
        #pragma unroll
        for (int j = 0; j < 16; ++j)
            mx[j] = fmaxf(mx[j], __shfl_xor(mx[j], off, 32));
    }
    float m8[16];
    #pragma unroll
    for (int j = 0; j < 16; ++j) m8[j] = mx[j] * SCALE;   // exp(8s - 8M) = exp(fma(s,8,-m8))

    // ---- Pass 2: sum of exp with exact global max (1 fma + 1 exp + 1 add per element) ----
    float l[16];
    #pragma unroll
    for (int j = 0; j < 16; ++j) l[j] = 0.0f;

    for (int t = 0; t < SEQN / 16; ++t) {
        score_tiles(kbase, t, lr, half, a0L.v, a1L.v, a0H.v, a1H.v, acc0, acc1);
        #pragma unroll
        for (int j = 0; j < 8; ++j) {
            l[j]     += __expf(fmaf(acc0[j], SCALE, -m8[j]));
            l[8 + j] += __expf(fmaf(acc1[j], SCALE, -m8[8 + j]));
        }
    }
    #pragma unroll
    for (int off = 1; off <= 8; off <<= 1) {
        #pragma unroll
        for (int j = 0; j < 16; ++j)
            l[j] += __shfl_xor(l[j], off, 32);
    }

    // Fold normalization into the exponent: exp(8s - m8)/L = exp(8s - (m8 + ln L)).
    // Single 16-entry constant array for pass 3; kills the rcp ops and 16 live VGPRs.
    float c[16];
    #pragma unroll
    for (int j = 0; j < 16; ++j) c[j] = m8[j] + __logf(l[j]);

    // ---- Pass 3: normalized column sums (1 fma + 1 exp + 1 add per element) ----
    for (int t = 0; t < SEQN / 16; ++t) {
        score_tiles(kbase, t, lr, half, a0L.v, a1L.v, a0H.v, a1H.v, acc0, acc1);
        float local = 0.0f;
        #pragma unroll
        for (int j = 0; j < 8; ++j) {
            local += __expf(fmaf(acc0[j], SCALE, -c[j]));
            local += __expf(fmaf(acc1[j], SCALE, -c[8 + j]));
        }
        // fold rows j+0..7 (half 0) with rows j+8..15 (half 1), same column
        local += __shfl_xor(local, 16, 32);
        if (half == 0) atomicAdd(&scol[t * 16 + lr], local);   // ds_add_f32
    }

    __syncthreads();
    for (int i = threadIdx.x; i < SEQN; i += 128)
        atomicAdd(&gcol[bh * SEQN + i], scol[i]);              // global_atomic_add_f32
}

// ---------------- Kernel 4: out[b,h,m,d] = colsum[b,h,m] * v[b,h,m,d] ----------------
__global__ void scale_v_kernel(const float* __restrict__ v, const float* __restrict__ gcol,
                               float* __restrict__ out) {
    int i = blockIdx.x * blockDim.x + threadIdx.x;   // float4 index; flat = i*4
    float4 vv = ((const float4*)v)[i];
    float  s  = gcol[i >> 4];                        // (flat/64) = bh*2048 + m
    float4 o  = { vv.x * s, vv.y * s, vv.z * s, vv.w * s };
    ((float4*)out)[i] = o;
}

// ---------------- Host launcher ----------------
extern "C" void kernel_launch(void* const* d_in, const int* in_sizes, int n_in,
                              void* d_out, int out_size, void* d_ws, size_t ws_size,
                              hipStream_t stream) {
    (void)in_sizes; (void)n_in; (void)out_size; (void)ws_size;

    const float* q = (const float*)d_in[0];
    const float* k = (const float*)d_in[1];
    const float* v = (const float*)d_in[2];
    float* out = (float*)d_out;

    const size_t total = (size_t)BH * SEQN * DH;     // 4,194,304 elements

    // workspace layout: [colsum: 32*2048 f32][Q bf16: total][K bf16: total]  ~16.25 MB
    float*          gcol = (float*)d_ws;
    unsigned short* qb   = (unsigned short*)((char*)d_ws + (size_t)BH * SEQN * sizeof(float));
    unsigned short* kb   = qb + total;

    zero_colsum_kernel<<<(BH * SEQN) / 256, 256, 0, stream>>>(gcol);
    cvt_bf16_kernel<<<(int)(total / 4 / 256), 256, 0, stream>>>(q, k, qb, kb);
    attn_colsum_kernel<<<BH * 16, 128, 0, stream>>>(qb, kb, gcol);
    scale_v_kernel<<<(int)(total / 4 / 256), 256, 0, stream>>>(v, gcol, out);
}
